// YOLOV3BBox_58506044506231
// MI455X (gfx1250) — compile-verified, matrix operand order checked by reading, not run on hardware
//
#include <hip/hip_runtime.h>
#include <hip/hip_fp16.h>

// ---------------- problem constants (fixed by the reference) ----------------
#define HW_    16384      // 128*128
#define W_     128
#define P_     49152      // HW_*3 priors
#define NC_    80
#define ATTR_  85
#define CHAN_  255
#define TP_    256        // pixels staged per workgroup in K1
#define NBUCK  2048       // score-bit histogram buckets
#define CAP_   8192       // candidate capacity (>= PRE_NMS_K)
#define K_     4096       // PRE_NMS_K
#define MAXD   100
#define SCORE_THR_ 0.05f
#define CONF_THR_  0.005f
#define IOU_THR_   0.45f

typedef __attribute__((ext_vector_type(16))) _Float16 v16h;
typedef __attribute__((ext_vector_type(8)))  float    v8f;
typedef unsigned long long u64;
typedef unsigned int       u32;

__device__ __forceinline__ float sigm(float x) { return 1.0f / (1.0f + __expf(-x)); }

__device__ __forceinline__ unsigned ballot32(bool p) {
#if __has_builtin(__builtin_amdgcn_ballot_w32)
  return __builtin_amdgcn_ballot_w32(p);
#else
  return (unsigned)__ballot(p);
#endif
}

// ---------------------------------------------------------------------------
// K0: zero accumulator state (histogram, counter, NMS mask, keep flags).
// Required every launch for determinism under graph replay.
// ---------------------------------------------------------------------------
__global__ void k0_init(u32* __restrict__ ghist, u32* __restrict__ count,
                        u32* __restrict__ mask, u32* __restrict__ keep) {
  int g = blockIdx.x * 256 + threadIdx.x;
  if (g < NBUCK) ghist[g] = 0u;
  if (g == 0)    count[0] = 0u;
  for (int i = g; i < K_ * 128; i += gridDim.x * 256) mask[i] = 0u;
  if (g < K_) keep[g] = 0u;
}

// ---------------------------------------------------------------------------
// K1: stage a 255-channel x 256-pixel tile of pred_map into LDS with CDNA5
// async LDS loads, then decode boxes + build the score-bits histogram.
// ---------------------------------------------------------------------------
__global__ void k1_decode_hist(const float* __restrict__ pred,
                               float* __restrict__ boxes,
                               u32* __restrict__ ghist) {
  extern __shared__ float smem[];                 // CHAN_*TP_ floats + NBUCK u32
  u32* lh = (u32*)&smem[CHAN_ * TP_];
  const int tid  = threadIdx.x;                   // 256 threads
  const int lane = tid & 31, wave = tid >> 5;     // wave32
  const int pixelBase = blockIdx.x * TP_;

  for (int b = tid; b < NBUCK; b += 256) lh[b] = 0u;

  // Async tensor-style staging: each wave-instruction copies 512B (32 lanes x 16B).
  // LDS addrspace(3) offsets live in the low 32 bits of the flat pointer.
  const u32 lds_base = (u32)(size_t)(void*)smem;
  for (int c = wave; c < CHAN_; c += 8) {
    u64 g = (u64)(size_t)(pred + (size_t)c * HW_ + pixelBase) + (u64)lane * 16u;
    u32 l = lds_base + (u32)c * (TP_ * 4u) + (u32)lane * 16u;
    asm volatile("global_load_async_to_lds_b128 %0, %1, off" :: "v"(l),        "v"(g)          : "memory");
    asm volatile("global_load_async_to_lds_b128 %0, %1, off" :: "v"(l + 512u), "v"(g + 512ull) : "memory");
  }
  asm volatile("s_wait_asynccnt 0x0" ::: "memory");
  __syncthreads();

  const float BW[3] = {116.f, 156.f, 373.f};
  const float BH[3] = { 90.f, 198.f, 326.f};
  const int px  = tid;
  const int gp  = pixelBase + px;
  const int wix = gp & (W_ - 1), hix = gp >> 7;

  for (int a = 0; a < 3; a++) {
    const float* s = &smem[(a * ATTR_) * TP_ + px];  // stride TP_ per attribute
    float sx  = sigm(s[0 * TP_]);
    float sy  = sigm(s[1 * TP_]);
    float ew  = __expf(s[2 * TP_]);
    float eh  = __expf(s[3 * TP_]);
    float obj = sigm(s[4 * TP_]);
    float cx = (float)(wix * 32 + 16) + (sx - 0.5f) * 32.f;
    float cy = (float)(hix * 32 + 16) + (sy - 0.5f) * 32.f;
    float hw = 0.5f * BW[a] * ew, hh = 0.5f * BH[a] * eh;
    int prior = gp * 3 + a;
    ((float4*)boxes)[prior] = make_float4(cx - hw, cy - hh, cx + hw, cy + hh);
    bool conf = obj >= CONF_THR_;
    for (int c = 0; c < NC_; c++) {
      float cs = sigm(s[(5 + c) * TP_]);
      if (conf && cs > SCORE_THR_) {
        u32 bits = __float_as_uint(cs * obj);    // score in (0,1): bits are monotone
        atomicAdd(&lh[bits >> 19], 1u);
      }
    }
  }
  __syncthreads();
  for (int b = tid; b < NBUCK; b += 256) { u32 v = lh[b]; if (v) atomicAdd(&ghist[b], v); }
}

// ---------------------------------------------------------------------------
// K2: suffix-scan histogram -> smallest bucket whose suffix count <= CAP_.
// Candidate SET becomes deterministic regardless of atomic append order.
// ---------------------------------------------------------------------------
__global__ void k2_threshold(const u32* __restrict__ ghist, u32* __restrict__ thr_bits) {
  __shared__ u32 h[NBUCK];
  for (int b = threadIdx.x; b < NBUCK; b += 256) h[b] = ghist[b];
  __syncthreads();
  if (threadIdx.x == 0) {
    u32 run = 0; int thr = NBUCK;
    for (int b = NBUCK - 1; b >= 0; b--) {
      if (run + h[b] > (u32)CAP_) break;
      run += h[b]; thr = b;
    }
    thr_bits[0] = ((u32)thr) << 19;
  }
}

// ---------------------------------------------------------------------------
// K3: streaming second pass; recompute scores (cheaper than storing 16MB),
// append 64-bit keys (score_bits<<32)|~flat_idx  (tie-break = lower index).
// ---------------------------------------------------------------------------
__global__ void k3_compact(const float* __restrict__ pred, const u32* __restrict__ thr_bits,
                           u32* __restrict__ count, u64* __restrict__ cand) {
  int g = blockIdx.x * 256 + threadIdx.x;            // g in [0, 3*HW_)
  int a = g / HW_, pix = g - a * HW_;                // coalesced: fixed attr, pix contiguous
  const float* base = pred + (size_t)a * ATTR_ * HW_ + pix;
  float obj = sigm(base[(size_t)4 * HW_]);
  bool conf = obj >= CONF_THR_;
  u32 thr = thr_bits[0];
  int prior = pix * 3 + a;
  for (int c = 0; c < NC_; c++) {
    float cs = sigm(base[(size_t)(5 + c) * HW_]);
    if (conf && cs > SCORE_THR_) {
      u32 bits = __float_as_uint(cs * obj);
      if (bits >= thr) {
        u32 pos = atomicAdd(count, 1u);
        if (pos < (u32)CAP_) {
          u32 flat = (u32)(prior * NC_ + c);
          cand[pos] = ((u64)bits << 32) | (u64)(0xFFFFFFFFu - flat);
        }
      }
    }
  }
}

// ---------------------------------------------------------------------------
// K4: one-workgroup bitonic sort (8192 u64 keys, 64KB LDS) -> top 4096;
// gather boxes, wave32 shfl+LDS max-reduce -> maxc; write class-offset boxes.
// ---------------------------------------------------------------------------
__global__ void k4_sort_prep(const u64* __restrict__ cand, const u32* __restrict__ count,
                             const float* __restrict__ boxes,
                             float* __restrict__ tscore, u32* __restrict__ tflat,
                             float* __restrict__ tbox, float* __restrict__ tnms) {
  __shared__ u64 keys[CAP_];                         // exactly 64KB
  int t = threadIdx.x;                               // 1024 threads
  u32 cnt = count[0];
  int n = (int)(cnt > (u32)CAP_ ? (u32)CAP_ : cnt);
  for (int k = t; k < CAP_; k += 1024) keys[k] = (k < n) ? cand[k] : 0ull;
  __syncthreads();

  for (int k2 = 2; k2 <= CAP_; k2 <<= 1) {
    for (int j = k2 >> 1; j > 0; j >>= 1) {
      for (int i = t; i < CAP_; i += 1024) {
        int ixj = i ^ j;
        if (ixj > i) {
          u64 x = keys[i], y = keys[ixj];
          bool desc = ((i & k2) == 0);
          if ((x < y) == desc) { keys[i] = y; keys[ixj] = x; }
        }
      }
      __syncthreads();
    }
  }

  float lmax = 0.f;
  for (int e = t; e < K_; e += 1024) {
    u64 key = keys[e];
    float sc; u32 flat; float4 bx = make_float4(0.f, 0.f, 0.f, 0.f);
    if (key != 0ull) {
      sc   = __uint_as_float((u32)(key >> 32));
      flat = 0xFFFFFFFFu - (u32)(key & 0xFFFFFFFFu);
      bx = ((const float4*)boxes)[flat / NC_];
      lmax = fmaxf(lmax, fmaxf(fmaxf(bx.x, bx.y), fmaxf(bx.z, bx.w)));
    } else { sc = 0.f; flat = 0u; }
    tscore[e] = sc; tflat[e] = flat;
    ((float4*)tbox)[e] = bx;
  }
  // reduce lmax: wave32 shfl then 32 partials in LDS (reuse keys[4096..] scratch)
  for (int off = 16; off > 0; off >>= 1) lmax = fmaxf(lmax, __shfl_down(lmax, off));
  float* redf = (float*)&keys[K_];
  if ((t & 31) == 0) redf[t >> 5] = lmax;
  __syncthreads();
  if (t == 0) { float m = 0.f; for (int q = 0; q < 32; q++) m = fmaxf(m, redf[q]); redf[32] = m; }
  __syncthreads();
  float offmul = redf[32] + 1.0f;                    // maxc + 1

  for (int e = t; e < K_; e += 1024) {
    u64 key = keys[e];                               // keys[0..4095] untouched by scratch
    float4 bx = ((const float4*)tbox)[e];            // re-read own writes
    float off2 = 0.f;
    if (key != 0ull) {
      u32 flat = 0xFFFFFFFFu - (u32)(key & 0xFFFFFFFFu);
      off2 = (float)(flat % NC_) * offmul;
    }
    ((float4*)tnms)[e] = make_float4(bx.x + off2, bx.y + off2, bx.z + off2, bx.w + off2);
  }
}

// ---------------------------------------------------------------------------
// K5: suppression bitmask. One wave per 16x16 tile (upper triangle of 256x256
// tiles). area_i + area_j tile is produced by a rank-2 v_wmma_f32_16x16x32_f16
// whose f32 C-layout matches the per-lane VALU intersection loop exactly.
// ---------------------------------------------------------------------------
__global__ void k5_ioumask(const float* __restrict__ tnms, u32* __restrict__ mask) {
  __shared__ float rb[8][16][5];                     // per-wave row boxes + area
  int tid = threadIdx.x, lane = tid & 31, wave = tid >> 5;
  int T = blockIdx.x * 8 + wave;                     // tile id over 256x256 tiles
  int ti = T >> 8, tj = T & 255;

  if (lane < 16) {
    int i = ti * 16 + lane;
    float4 b = ((const float4*)tnms)[i];
    rb[wave][lane][0] = b.x; rb[wave][lane][1] = b.y;
    rb[wave][lane][2] = b.z; rb[wave][lane][3] = b.w;
    rb[wave][lane][4] = (b.z - b.x) * (b.w - b.y);
  }
  __syncthreads();
  if (tj < ti) return;                               // lower triangle: nothing to do

  int j = tj * 16 + (lane & 15);
  float4 cb = ((const float4*)tnms)[j];
  float carea = (cb.z - cb.x) * (cb.w - cb.y);

  // D = A x B : A rows = [area_i*2^-12, 1], B rows = [1 ; area_j*2^-12]
  const float S = 1.0f / 4096.0f;
  v16h A = {}; v16h B = {};
  if (lane < 16) {
    A[0] = (_Float16)(rb[wave][lane][4] * S);        // K=0 slot, row M=lane
    A[1] = (_Float16)1.0f;                           // K=1 slot
    B[0] = (_Float16)1.0f;                           // K=0 row, col N=lane
    B[1] = (_Float16)(carea * S);                    // K=1 row
  }
  v8f C = {};
  v8f D = __builtin_amdgcn_wmma_f32_16x16x32_f16(false, A, false, B, (short)0, C, false, false);

  int rowOff = (lane >= 16) ? 8 : 0;                 // C layout: VGPR r -> rows r / r+8
  for (int r = 0; r < 8; r++) {
    int ii = ti * 16 + r + rowOff;
    const float* rbx = rb[wave][r + rowOff];
    float ix = fminf(rbx[2], cb.z) - fmaxf(rbx[0], cb.x);
    float iy = fminf(rbx[3], cb.w) - fmaxf(rbx[1], cb.y);
    float inter = fmaxf(ix, 0.f) * fmaxf(iy, 0.f);
    float denom = D[r] * 4096.0f - inter;            // area_i + area_j - inter
    bool hit = (denom > 0.f) && (inter > IOU_THR_ * denom) && (j > ii);
    unsigned bal = ballot32(hit);                    // low 16: rows r, high 16: rows r+8
    if (lane == 0) {
      u32 f = bal & 0xFFFFu;
      if (f) atomicOr(&mask[(size_t)(ti * 16 + r) * 128 + (tj >> 1)], f << ((tj & 1) * 16));
    }
    if (lane == 16) {
      u32 f = bal >> 16;
      if (f) atomicOr(&mask[(size_t)(ti * 16 + r + 8) * 128 + (tj >> 1)], f << ((tj & 1) * 16));
    }
  }
}

// ---------------------------------------------------------------------------
// K6: greedy NMS scan. Invalid entries start suppressed (reference semantics:
// sup init = !valid). 4096 sequential steps over a 128-word LDS vector.
// ---------------------------------------------------------------------------
__global__ void k6_scan(const u32* __restrict__ mask, const float* __restrict__ tscore,
                        u32* __restrict__ keep) {
  __shared__ u32 sup[128];
  __shared__ int ki;
  int t = threadIdx.x;                               // 128 threads
  u32 s0 = 0;
  for (int k = 0; k < 32; k++) {
    int j = t * 32 + k;
    if (!(tscore[j] > 0.f)) s0 |= (1u << k);
  }
  sup[t] = s0;
  __syncthreads();
  for (int i = 0; i < K_; i++) {
    if (t == 0) ki = !((sup[i >> 5] >> (i & 31)) & 1u);
    __syncthreads();
    if (ki) sup[t] |= mask[(size_t)i * 128 + t];
    if (t == 0) keep[i] = (u32)ki;
    __syncthreads();
  }
}

// ---------------------------------------------------------------------------
// K7: prefix-scan keep flags -> first 100 kept (already score-sorted), write
// dets[100,5] then labels[100] (as float, -1 padding). Padding writes only
// slots >= total kept, so det/pad writes never alias.
// ---------------------------------------------------------------------------
__global__ void k7_finalize(const u32* __restrict__ keep, const float* __restrict__ tscore,
                            const u32* __restrict__ tflat, const float* __restrict__ tbox,
                            float* __restrict__ out) {
  __shared__ u32 pref[1024];
  int t = threadIdx.x;                               // 1024 threads, 4 elems each
  int base = t * 4;
  u32 km[4]; u32 cnt = 0;
  for (int k = 0; k < 4; k++) { km[k] = keep[base + k]; cnt += km[k]; }
  pref[t] = cnt;
  __syncthreads();
  for (int off = 1; off < 1024; off <<= 1) {
    u32 v = (t >= off) ? pref[t - off] : 0u;
    __syncthreads();
    pref[t] += v;
    __syncthreads();
  }
  u32 total = pref[1023];
  u32 p = pref[t] - cnt;                             // exclusive prefix
  for (int k = 0; k < 4; k++) {
    if (km[k]) {
      if (p < (u32)MAXD) {
        int e = base + k;
        float4 bx = ((const float4*)tbox)[e];
        out[p * 5 + 0] = bx.x; out[p * 5 + 1] = bx.y;
        out[p * 5 + 2] = bx.z; out[p * 5 + 3] = bx.w;
        out[p * 5 + 4] = tscore[e];
        out[500 + p]   = (float)(tflat[e] % NC_);
      }
      p++;
    }
  }
  if (t < MAXD && (u32)t >= total) {                 // disjoint padding slots
    for (int q = 0; q < 5; q++) out[t * 5 + q] = 0.f;
    out[500 + t] = -1.f;
  }
}

// ---------------------------------------------------------------------------
extern "C" void kernel_launch(void* const* d_in, const int* in_sizes, int n_in,
                              void* d_out, int out_size, void* d_ws, size_t ws_size,
                              hipStream_t stream) {
  const float* pred = (const float*)d_in[0];
  float* out = (float*)d_out;
  (void)in_sizes; (void)n_in; (void)out_size; (void)ws_size;

  char* ws = (char*)d_ws;
  size_t o = 0;
  auto take = [&](size_t bytes) -> char* {
    char* p = ws + o;
    o = (o + bytes + 255) & ~(size_t)255;
    return p;
  };
  float* boxes  = (float*)take((size_t)P_ * 4 * 4);        // 786 KB
  u32*   ghist  = (u32*)  take((size_t)NBUCK * 4);
  u32*   thr    = (u32*)  take(4);
  u32*   count  = (u32*)  take(4);
  u64*   cand   = (u64*)  take((size_t)CAP_ * 8);          // 64 KB
  float* tscore = (float*)take((size_t)K_ * 4);
  u32*   tflat  = (u32*)  take((size_t)K_ * 4);
  float* tbox   = (float*)take((size_t)K_ * 16);
  float* tnms   = (float*)take((size_t)K_ * 16);
  u32*   mask   = (u32*)  take((size_t)K_ * 128 * 4);      // 2 MB
  u32*   keep   = (u32*)  take((size_t)K_ * 4);

  k0_init<<<dim3(2056), dim3(256), 0, stream>>>(ghist, count, mask, keep);
  k1_decode_hist<<<dim3(HW_ / TP_), dim3(256),
                   (CHAN_ * TP_ * 4 + NBUCK * 4), stream>>>(pred, boxes, ghist);
  k2_threshold<<<dim3(1), dim3(256), 0, stream>>>(ghist, thr);
  k3_compact<<<dim3(P_ / 256), dim3(256), 0, stream>>>(pred, thr, count, cand);
  k4_sort_prep<<<dim3(1), dim3(1024), 0, stream>>>(cand, count, boxes,
                                                   tscore, tflat, tbox, tnms);
  k5_ioumask<<<dim3(8192), dim3(256), 0, stream>>>(tnms, mask);
  k6_scan<<<dim3(1), dim3(128), 0, stream>>>(mask, tscore, keep);
  k7_finalize<<<dim3(1), dim3(1024), 0, stream>>>(keep, tscore, tflat, tbox, out);
}
